// NonLocal_11441792877327
// MI455X (gfx1250) — matrix operand order repaired and checked by reading, not compiled, not run on hardware
//
#include <hip/hip_runtime.h>

// ---------------------------------------------------------------------------
// NonLocal attention block for MI455X (gfx1250), wave32 + WMMA bf16.
//   B=8, C=256, C'=32, N=4096.  All GEMMs: v_wmma_f32_16x16x32_bf16 (K=32).
//   Flash-style online softmax on S^T so each lane owns one softmax row:
//   row stats are per-lane scalars, reductions are 1 shfl_xor(16) each.
// ---------------------------------------------------------------------------

typedef __attribute__((ext_vector_type(8)))  float        v8f;
typedef __attribute__((ext_vector_type(4)))  float        v4f;
typedef __attribute__((ext_vector_type(4)))  unsigned int v4u;
typedef __attribute__((ext_vector_type(16))) __bf16       v16bf;
typedef __attribute__((ext_vector_type(2)))  __bf16       v2bf;

union BF16x16 {
    v16bf          v;
    unsigned short h[16];
    unsigned int   u[8];
    v4u            q[2];
};

__device__ __forceinline__ unsigned short f32_to_bf16(float f) {
    union { float f; unsigned u; } cv; cv.f = f;
    unsigned u = cv.u;
    u += 0x7FFFu + ((u >> 16) & 1u);           // round-to-nearest-even
    return (unsigned short)(u >> 16);
}

// Pack two f32 -> packed bf16 pair (lo in [15:0], hi in [31:16]), RNE.
// CDNA5 has V_CVT_PK_BF16_F32 but the clang builtin is not exposed on this
// toolchain -> use inline asm (probe-verified by this compile loop).
__device__ __forceinline__ unsigned int pk_bf16(float lo, float hi) {
#if __has_builtin(__builtin_amdgcn_cvt_pk_bf16_f32)
    union { v2bf v; unsigned u; } cv;
    cv.v = __builtin_amdgcn_cvt_pk_bf16_f32(lo, hi);
    return cv.u;
#else
    unsigned int r;
    asm("v_cvt_pk_bf16_f32 %0, %1, %2" : "=v"(r) : "v"(lo), "v"(hi));
    return r;
#endif
}

__device__ __forceinline__ float fast_exp2(float v) {
#if __has_builtin(__builtin_amdgcn_exp2f)
    return __builtin_amdgcn_exp2f(v);          // bare v_exp_f32
#else
    return exp2f(v);
#endif
}

// A-operand (16x32, 16-bit) K index for half h in lane L:
//   lanes 0-15: h0..7 -> K0..7,  h8..15 -> K16..23
//   lanes16-31: h0..7 -> K8..15, h8..15 -> K24..31
__device__ __forceinline__ int a_k(int h, int lh) { return h + 8 * ((h >> 3) + lh); }

#define LOG2E 1.4426950408889634f

// ---------------------------------------------------------------------------
// Kernel 0: pre-swizzle weights to bf16 WMMA operand layouts.
//  wB  [3][8][2][32][16] : w_theta/w_phi/w_g as (K=c 32) x (c' 16) B tiles
//  woA [16][32][16]      : w_out as A tiles (rows c 16, K=c' 32)
// ---------------------------------------------------------------------------
__global__ void prep_weights(const float* __restrict__ wt,
                             const float* __restrict__ wp,
                             const float* __restrict__ wg,
                             const float* __restrict__ wo,
                             unsigned short* __restrict__ wB,
                             unsigned short* __restrict__ woA) {
    int tid = blockIdx.x * blockDim.x + threadIdx.x;
    if (tid < 3 * 8 * 2 * 32 * 16) {
        int h    = tid & 15;
        int lane = (tid >> 4) & 31;
        int t    = (tid >> 9) & 1;
        int kk   = (tid >> 10) & 7;
        int w    = tid >> 13;                       // 0..2
        int cp   = 16 * t + (lane & 15);            // c' (B column)
        int c    = 32 * kk + a_k(h, lane >> 4);     // c  (K)
        const float* src = (w == 0) ? wt : (w == 1) ? wp : wg;
        wB[tid] = f32_to_bf16(src[cp * 256 + c]);
    } else {
        int t2   = tid - 3 * 8 * 2 * 32 * 16;       // [0, 16*32*16)
        int h    = t2 & 15;
        int lane = (t2 >> 4) & 31;
        int ct   = t2 >> 9;                         // 0..15
        int c    = 16 * ct + (lane & 15);           // A row (output channel)
        int k    = a_k(h, lane >> 4);               // K = c'
        woA[t2] = f32_to_bf16(wo[c * 32 + k]);
    }
}

// ---------------------------------------------------------------------------
// Kernel 1: projections theta/phi/g via WMMA; outputs pre-swizzled:
//  thetaB[b][nt][lane][16] : B-layout cols n, K=c'   (theta scaled by log2e)
//  phiA  [b][nt][lane][16] : A-layout rows m,  K=c'
//  gA    [b][nt][f][lane][8]: half A-operand, [lane][e] = g(16nt+e+8lh,16f+l15)
//                             == direct copy of this lane's C-layout regs.
// ---------------------------------------------------------------------------
__global__ __launch_bounds__(32)
void proj_kernel(const float* __restrict__ x,
                 const float* __restrict__ bth, const float* __restrict__ bph,
                 const float* __restrict__ bgg,
                 const unsigned short* __restrict__ wB,
                 unsigned short* __restrict__ thetaB,
                 unsigned short* __restrict__ phiA,
                 unsigned short* __restrict__ gA) {
    __shared__ __align__(16) unsigned short scr[16][40];   // 80B row stride
    const int b   = blockIdx.x >> 8;
    const int nt  = blockIdx.x & 255;
    const int n0  = nt * 16;
    const int lane = threadIdx.x, l15 = lane & 15, lh = lane >> 4;

    v8f acc[6];
#pragma unroll
    for (int i = 0; i < 6; ++i) acc[i] = (v8f)0.0f;

    for (int kk = 0; kk < 8; ++kk) {
        float xv[16];
#pragma unroll
        for (int h = 0; h < 16; ++h) {
            int K = a_k(h, lh);
            xv[h] = x[(size_t)(b * 256 + kk * 32 + K) * 4096 + n0 + l15];
        }
        BF16x16 A;
#pragma unroll
        for (int k = 0; k < 8; ++k)
            A.u[k] = pk_bf16(xv[2 * k], xv[2 * k + 1]);   // v_cvt_pk_bf16_f32
#pragma unroll
        for (int w6 = 0; w6 < 6; ++w6) {   // w6 = matrix*2 + c'-tile
            const v16bf* bp = (const v16bf*)(wB +
                ((size_t)((((w6 >> 1) * 8 + kk) * 2 + (w6 & 1)) * 32 + lane) * 16));
            acc[w6] = __builtin_amdgcn_wmma_f32_16x16x32_bf16(
                false, A.v, false, *bp, (short)0, acc[w6], false, false);
        }
    }

    const float bias[6] = { bth[l15], bth[16 + l15],
                            bph[l15], bph[16 + l15],
                            bgg[l15], bgg[16 + l15] };

    // ---- theta (pre-scaled by log2e) -> B-layout ----------------------
#pragma unroll
    for (int t = 0; t < 2; ++t)
#pragma unroll
        for (int j = 0; j < 8; ++j)
            scr[j + 8 * lh][l15 + 16 * t] =
                f32_to_bf16((acc[t][j] + bias[t]) * LOG2E);
    __syncthreads();
    {
        BF16x16 o;
        o.q[0] = *(const v4u*)&scr[l15][8 * lh];
        o.q[1] = *(const v4u*)&scr[l15][16 + 8 * lh];
        *(v16bf*)(thetaB + ((size_t)((b * 256 + nt) * 32 + lane) * 16)) = o.v;
    }
    __syncthreads();

    // ---- phi -> A-layout ----------------------------------------------
#pragma unroll
    for (int t = 0; t < 2; ++t)
#pragma unroll
        for (int j = 0; j < 8; ++j)
            scr[j + 8 * lh][l15 + 16 * t] = f32_to_bf16(acc[2 + t][j] + bias[2 + t]);
    __syncthreads();
    {
        BF16x16 o;
        o.q[0] = *(const v4u*)&scr[l15][8 * lh];        // K = 0..7 (+8*lh)
        o.q[1] = *(const v4u*)&scr[l15][16 + 8 * lh];   // K = 16..23 (+8*lh)
        *(v16bf*)(phiA + ((size_t)((b * 256 + nt) * 32 + lane) * 16)) = o.v;
    }

    // ---- g -> half A-operand: identity on this lane's registers --------
#pragma unroll
    for (int f = 0; f < 2; ++f) {
        union { v4u q; unsigned int u[4]; } o;
#pragma unroll
        for (int k = 0; k < 4; ++k)
            o.u[k] = pk_bf16(acc[4 + f][2 * k]     + bias[4 + f],
                             acc[4 + f][2 * k + 1] + bias[4 + f]);
        *(v4u*)(gA + ((size_t)(((b * 256 + nt) * 2 + f) * 32 + lane) * 8)) = o.q;
    }
}

// ---------------------------------------------------------------------------
// Kernel 2: flash attention on S^T + output projection + residual.
//   One wave per (b, 16-row tile). Lane owns softmax row n = l15 (both halves
//   combined with a single shfl_xor(16)).
// ---------------------------------------------------------------------------
__global__ __launch_bounds__(32)
void attn_kernel(const float* __restrict__ x,
                 const float* __restrict__ bo,
                 const float* __restrict__ gamma,
                 const unsigned short* __restrict__ thetaB,
                 const unsigned short* __restrict__ phiA,
                 const unsigned short* __restrict__ gA,
                 const unsigned short* __restrict__ woA,
                 float* __restrict__ out) {
    __shared__ __align__(16) unsigned char smem[1280];
    unsigned short (*scr)[40] = (unsigned short(*)[40])smem;   // 16 x 40 halves
    float          (*scrF)[20] = (float(*)[20])smem;           // 16 x 20 floats

    const int b   = blockIdx.x >> 8;
    const int rt  = blockIdx.x & 255;
    const int n0  = rt * 16;
    const int lane = threadIdx.x, l15 = lane & 15, lh = lane >> 4;

    // loop-invariant theta^T B operand (cols n of this tile)
    const v16bf tb = *(const v16bf*)(thetaB +
        ((size_t)((b * 256 + rt) * 32 + lane) * 16));

    v8f y0 = (v8f)0.0f, y1 = (v8f)0.0f;   // Y^T: rows c' 0..15 / 16..31, col n=l15
    float mr = -1e30f, lr = 0.0f;
    const v8f zc = (v8f)0.0f;

    for (int it = 0; it < 128; ++it) {
        // S^T subtiles: rows m, cols n
        const v16bf pa0 = *(const v16bf*)(phiA +
            ((size_t)((b * 256 + 2 * it) * 32 + lane) * 16));
        const v16bf pa1 = *(const v16bf*)(phiA +
            ((size_t)((b * 256 + 2 * it + 1) * 32 + lane) * 16));
        v8f s0 = __builtin_amdgcn_wmma_f32_16x16x32_bf16(false, pa0, false, tb,
                                                         (short)0, zc, false, false);
        v8f s1 = __builtin_amdgcn_wmma_f32_16x16x32_bf16(false, pa1, false, tb,
                                                         (short)0, zc, false, false);
        // online softmax (log2 domain); lane owns column n = l15
        float sm = -1e30f;
#pragma unroll
        for (int j = 0; j < 8; ++j) {
            sm = fmaxf(sm, s0[j]);
            sm = fmaxf(sm, s1[j]);
        }
        sm = fmaxf(sm, __shfl_xor(sm, 16, 32));
        float mn = fmaxf(mr, sm);
        float al = fast_exp2(mr - mn);
        mr = mn;
        float p0[8], p1[8], rs = 0.0f;
#pragma unroll
        for (int j = 0; j < 8; ++j) {
            p0[j] = fast_exp2(s0[j] - mn);
            p1[j] = fast_exp2(s1[j] - mn);
            rs += p0[j] + p1[j];
        }
        rs += __shfl_xor(rs, 16, 32);
        lr = lr * al + rs;
        y0 *= al;
        y1 *= al;
        // stage P^T into LDS: scr[n][m], rows n, 32 m-cols contiguous
#pragma unroll
        for (int k = 0; k < 4; ++k) {
            *(unsigned int*)&scr[l15][8 * lh + 2 * k]      = pk_bf16(p0[2*k], p0[2*k+1]);
            *(unsigned int*)&scr[l15][16 + 8 * lh + 2 * k] = pk_bf16(p1[2*k], p1[2*k+1]);
        }
        __syncthreads();
        BF16x16 pb;                         // P^T as B operand (K=m, col n)
        pb.q[0] = *(const v4u*)&scr[l15][8 * lh];
        pb.q[1] = *(const v4u*)&scr[l15][16 + 8 * lh];
        // g^T A operands: halves 0..7 from even ntile, 8..15 from odd
        size_t ga0 = ((size_t)(((b * 256 + 2 * it) * 2) * 32 + lane)) * 8;
        BF16x16 g0, g1;
        g0.q[0] = *(const v4u*)(gA + ga0);
        g0.q[1] = *(const v4u*)(gA + ga0 + 512);
        g1.q[0] = *(const v4u*)(gA + ga0 + 256);
        g1.q[1] = *(const v4u*)(gA + ga0 + 768);
        __syncthreads();
        y0 = __builtin_amdgcn_wmma_f32_16x16x32_bf16(false, g0.v, false, pb.v,
                                                     (short)0, y0, false, false);
        y1 = __builtin_amdgcn_wmma_f32_16x16x32_bf16(false, g1.v, false, pb.v,
                                                     (short)0, y1, false, false);
    }

    // normalize, stage Y^T as B operand (K=c', col n) via LDS
    float inv = 1.0f / lr;
#pragma unroll
    for (int k = 0; k < 4; ++k) {
        *(unsigned int*)&scr[l15][8 * lh + 2 * k] =
            pk_bf16(y0[2 * k] * inv, y0[2 * k + 1] * inv);
        *(unsigned int*)&scr[l15][16 + 8 * lh + 2 * k] =
            pk_bf16(y1[2 * k] * inv, y1[2 * k + 1] * inv);
    }
    __syncthreads();
    BF16x16 yb;
    yb.q[0] = *(const v4u*)&scr[l15][8 * lh];
    yb.q[1] = *(const v4u*)&scr[l15][16 + 8 * lh];
    __syncthreads();

    const float gm = gamma[0];
    for (int ct = 0; ct < 16; ++ct) {
        v16bf wa = *(const v16bf*)(woA + ((size_t)(ct * 32 + lane) * 16));
        v8f o = __builtin_amdgcn_wmma_f32_16x16x32_bf16(false, wa, false, yb.v,
                                                        (short)0, zc, false, false);
        // transpose 16x16 (c-rows x n-cols) -> per-lane contiguous n for b128 I/O
#pragma unroll
        for (int j = 0; j < 8; ++j) scrF[j + 8 * lh][l15] = o[j];
        __syncthreads();
        int c = ct * 16 + l15;
        float bias = bo[c];
        size_t base = (size_t)(b * 256 + c) * 4096 + n0 + 8 * lh;
        v4f a0 = *(const v4f*)&scrF[l15][8 * lh];
        v4f a1 = *(const v4f*)&scrF[l15][8 * lh + 4];
        v4f xv0 = *(const v4f*)(x + base);
        v4f xv1 = *(const v4f*)(x + base + 4);
        v4f r0, r1;
#pragma unroll
        for (int j = 0; j < 4; ++j) {
            r0[j] = gm * (a0[j] + bias) + xv0[j];
            r1[j] = gm * (a1[j] + bias) + xv1[j];
        }
        *(v4f*)(out + base)     = r0;
        *(v4f*)(out + base + 4) = r1;
        __syncthreads();
    }
}

// ---------------------------------------------------------------------------
extern "C" void kernel_launch(void* const* d_in, const int* in_sizes, int n_in,
                              void* d_out, int out_size, void* d_ws, size_t ws_size,
                              hipStream_t stream) {
    (void)in_sizes; (void)n_in; (void)out_size; (void)ws_size;
    const float* x       = (const float*)d_in[0];
    const float* w_theta = (const float*)d_in[1];
    const float* b_theta = (const float*)d_in[2];
    const float* w_phi   = (const float*)d_in[3];
    const float* b_phi   = (const float*)d_in[4];
    const float* w_g     = (const float*)d_in[5];
    const float* b_g     = (const float*)d_in[6];
    const float* w_out   = (const float*)d_in[7];
    const float* b_out   = (const float*)d_in[8];
    const float* gamma   = (const float*)d_in[9];

    char* ws = (char*)d_ws;
    unsigned short* thetaB = (unsigned short*)(ws);                             // 2 MB
    unsigned short* phiA   = (unsigned short*)(ws + (size_t)2 * 1024 * 1024);   // 2 MB
    unsigned short* gA     = (unsigned short*)(ws + (size_t)4 * 1024 * 1024);   // 2 MB
    unsigned short* wB     = (unsigned short*)(ws + (size_t)6 * 1024 * 1024);   // 48 KB
    unsigned short* woA    = (unsigned short*)(ws + (size_t)6 * 1024 * 1024 + 49152); // 16 KB

    prep_weights<<<128, 256, 0, stream>>>(w_theta, w_phi, w_g, w_out, wB, woA);
    proj_kernel<<<2048, 32, 0, stream>>>(x, b_theta, b_phi, b_g, wB,
                                         thetaB, phiA, gA);
    attn_kernel<<<2048, 32, 0, stream>>>(x, b_out, gamma,
                                         thetaB, phiA, gA, woA, (float*)d_out);
}
